// MultitaskExtendedCTM_34892314313366
// MI455X (gfx1250) — matrix operand order repaired and checked by reading, not compiled, not run on hardware
//
#include <hip/hip_runtime.h>

// ---------------------------------------------------------------------------
// MI455X (gfx1250) implementation of MultitaskExtendedCTM forward.
// All dense layers run on v_wmma_f32_16x16x32_bf16 (wave32 WMMA).
// Each wave computes a 16x64 output strip: A fragment converted once per
// k-step and reused by 4 consecutive WMMAs (amortizes f32->bf16 cvt 4x).
// ---------------------------------------------------------------------------

typedef __attribute__((ext_vector_type(16))) __bf16        v16bf;
typedef __attribute__((ext_vector_type(8)))  float         v8f;
typedef __attribute__((ext_vector_type(4)))  float         f32x4;
typedef __attribute__((ext_vector_type(4)))  unsigned int  u32x4;

__device__ __forceinline__ unsigned short f2bf(float f) {
  union { float f; unsigned int u; } c; c.f = f;
  unsigned int u = c.u;
  unsigned int r = u + 0x7FFFu + ((u >> 16) & 1u);   // round-to-nearest-even
  return (unsigned short)(r >> 16);
}

// ---------------------------------------------------------------------------
// Repack f32 weights (K x N, row-major, x@W convention) into bf16 WMMA
// B-fragment-contiguous layout: tile = kt*NT + nt, then [lane][j] with
// k = kt*32 + 16*(lane>>4) + j, n = nt*16 + (lane&15).  512 ushorts / tile.
// NT may be padded (n >= Nreal packs zeros).
// ---------------------------------------------------------------------------
__global__ void pack_b_kernel(const float* __restrict__ W, long wBatch, int ldb,
                              int K, int Nreal, int NT,
                              unsigned short* __restrict__ out, long oBatch)
{
  int t = blockIdx.x * 256 + threadIdx.x;
  int KT = K >> 5;
  int total = KT * NT * 32;
  if (t >= total) return;
  int lane  = t & 31;
  int tile  = t >> 5;
  int nt    = tile % NT;
  int n     = nt * 16 + (lane & 15);
  int kbase = (tile / NT) * 32 + 16 * (lane >> 4);
  const float* Wz = W + (long)blockIdx.y * wBatch;
  unsigned short* o = out + (long)blockIdx.y * oBatch + (long)t * 16;
  for (int j = 0; j < 16; ++j) {
    float v = (n < Nreal) ? Wz[(long)(kbase + j) * ldb + n] : 0.0f;
    o[j] = f2bf(v);
  }
}

// ---------------------------------------------------------------------------
// Generic (optionally batched) GEMM: C[z] = A[z] (64 x K, f32 row-major)
//                                        * Bpack[z] (K x N, bf16 frag layout)
//                                        + bias[z]
// One wave per 16x64 output strip (4 N-tiles); K in steps of 32 via WMMA.
// NT must be a multiple of 4 (pack is zero-padded past Nreal).
// ---------------------------------------------------------------------------
__global__ __launch_bounds__(32) void gemm_bf16_wmma(
    const float* __restrict__ A, int lda, long aBatch,
    const unsigned short* __restrict__ Bp, long bBatch,
    int KT, int NT,
    float* __restrict__ C, int ldc, long cBatch,
    const float* __restrict__ bias, long biasBatch, int Nreal)
{
  const int lane = threadIdx.x & 31;
  const int nt0  = blockIdx.x * 4;          // first of 4 N-tiles
  const int mt   = blockIdx.y;
  const int z    = blockIdx.z;
  const int half = lane >> 4;
  const int mrow = lane & 15;

  const float* Ab = A + (long)z * aBatch + (long)(mt * 16 + mrow) * lda + 16 * half;
  const unsigned short* Bb = Bp + (long)z * bBatch;

  v8f acc0 = {0.f, 0.f, 0.f, 0.f, 0.f, 0.f, 0.f, 0.f};
  v8f acc1 = acc0, acc2 = acc0, acc3 = acc0;

  for (int kt = 0; kt < KT; ++kt) {
    // A fragment: row = mrow, k = kt*32 + 16*half + {0..15}
    const float* p0 = Ab + kt * 32;
    f32x4 a0 = *(const f32x4*)(p0);
    f32x4 a1 = *(const f32x4*)(p0 + 4);
    f32x4 a2 = *(const f32x4*)(p0 + 8);
    f32x4 a3 = *(const f32x4*)(p0 + 12);
    union { unsigned short s[16]; v16bf v; } au;
    au.s[0]  = f2bf(a0[0]); au.s[1]  = f2bf(a0[1]); au.s[2]  = f2bf(a0[2]); au.s[3]  = f2bf(a0[3]);
    au.s[4]  = f2bf(a1[0]); au.s[5]  = f2bf(a1[1]); au.s[6]  = f2bf(a1[2]); au.s[7]  = f2bf(a1[3]);
    au.s[8]  = f2bf(a2[0]); au.s[9]  = f2bf(a2[1]); au.s[10] = f2bf(a2[2]); au.s[11] = f2bf(a2[3]);
    au.s[12] = f2bf(a3[0]); au.s[13] = f2bf(a3[1]); au.s[14] = f2bf(a3[2]); au.s[15] = f2bf(a3[3]);

    // 4 B fragments (contiguous 32B per lane each), 4 back-to-back WMMAs
    const unsigned short* bp = Bb + ((long)(kt * NT + nt0) * 32 + lane) * 16;
    union { u32x4 q[2]; v16bf v; } b0, b1, b2, b3;
    b0.q[0] = *(const u32x4*)(bp);             b0.q[1] = *(const u32x4*)(bp + 8);
    b1.q[0] = *(const u32x4*)(bp + 512);       b1.q[1] = *(const u32x4*)(bp + 520);
    b2.q[0] = *(const u32x4*)(bp + 1024);      b2.q[1] = *(const u32x4*)(bp + 1032);
    b3.q[0] = *(const u32x4*)(bp + 1536);      b3.q[1] = *(const u32x4*)(bp + 1544);
    if (kt + 1 < KT)
      __builtin_prefetch((const void*)(bp + (long)NT * 512), 0, 1);

    acc0 = __builtin_amdgcn_wmma_f32_16x16x32_bf16(false, au.v, false, b0.v,
                                                   (short)0, acc0, false, false);
    acc1 = __builtin_amdgcn_wmma_f32_16x16x32_bf16(false, au.v, false, b1.v,
                                                   (short)0, acc1, false, false);
    acc2 = __builtin_amdgcn_wmma_f32_16x16x32_bf16(false, au.v, false, b2.v,
                                                   (short)0, acc2, false, false);
    acc3 = __builtin_amdgcn_wmma_f32_16x16x32_bf16(false, au.v, false, b3.v,
                                                   (short)0, acc3, false, false);
  }

  const float* biasb = bias ? bias + (long)z * biasBatch : nullptr;
  float* Cz = C + (long)z * cBatch;
#pragma unroll
  for (int i = 0; i < 4; ++i) {
    const v8f acc = (i == 0) ? acc0 : (i == 1) ? acc1 : (i == 2) ? acc2 : acc3;
    const int n = (nt0 + i) * 16 + mrow;
    if (n < Nreal) {
      const float bv = biasb ? biasb[n] : 0.0f;
      float* Cb = Cz + n;
#pragma unroll
      for (int r = 0; r < 8; ++r) {
        const int m = mt * 16 + r + 8 * half;
        Cb[(long)m * ldc] = acc[r] + bv;
      }
    }
  }
}

// ------------------------------ CNN encoder --------------------------------
__global__ void conv3x3_kernel(const float* __restrict__ in, const float* __restrict__ w,
                               const float* __restrict__ bias, float* __restrict__ out,
                               int Ci, int Co, int H, int W, int total)
{
  int idx = blockIdx.x * 256 + threadIdx.x;
  if (idx >= total) return;
  int x  = idx % W;
  int y  = (idx / W) % H;
  int co = (idx / (W * H)) % Co;
  int b  = idx / (W * H * Co);
  float acc = bias[co];
  for (int ci = 0; ci < Ci; ++ci) {
    const float* ip = in + (long)(b * Ci + ci) * H * W;
    const float* wp = w + (long)(co * Ci + ci) * 9;
    for (int ky = 0; ky < 3; ++ky) {
      int yy = y + ky - 1;
      if (yy < 0 || yy >= H) continue;
      for (int kx = 0; kx < 3; ++kx) {
        int xx = x + kx - 1;
        if (xx < 0 || xx >= W) continue;
        acc += ip[yy * W + xx] * wp[ky * 3 + kx];
      }
    }
  }
  out[idx] = acc;
}

__global__ void bn_stats_kernel(const float* __restrict__ buf, int C, int HW,
                                const float* __restrict__ g, const float* __restrict__ bta,
                                float* __restrict__ stats)
{
  int c = blockIdx.x;
  __shared__ float s1[256], s2[256];
  float sum = 0.f, ss = 0.f;
  int n = 64 * HW;
  for (int i = threadIdx.x; i < n; i += 256) {
    int b = i / HW, p = i % HW;
    float v = buf[(long)(b * C + c) * HW + p];
    sum += v; ss += v * v;
  }
  s1[threadIdx.x] = sum; s2[threadIdx.x] = ss;
  __syncthreads();
  for (int w = 128; w > 0; w >>= 1) {
    if (threadIdx.x < w) { s1[threadIdx.x] += s1[threadIdx.x + w]; s2[threadIdx.x] += s2[threadIdx.x + w]; }
    __syncthreads();
  }
  if (threadIdx.x == 0) {
    float mean = s1[0] / n;
    float var  = s2[0] / n - mean * mean;
    float sc   = g[c] * rsqrtf(var + 1e-5f);
    stats[c * 2]     = sc;
    stats[c * 2 + 1] = bta[c] - mean * sc;
  }
}

__global__ void bn_apply_relu_kernel(float* __restrict__ buf, const float* __restrict__ stats,
                                     int C, int HW, int total)
{
  int idx = blockIdx.x * 256 + threadIdx.x;
  if (idx >= total) return;
  int c = (idx / HW) % C;
  float v = buf[idx] * stats[c * 2] + stats[c * 2 + 1];
  buf[idx] = fmaxf(v, 0.f);
}

__global__ void maxpool2_kernel(const float* __restrict__ in, float* __restrict__ out,
                                int C, int H, int W, int total)
{
  int idx = blockIdx.x * 256 + threadIdx.x;
  if (idx >= total) return;
  int Ho = H / 2, Wo = W / 2;
  int x = idx % Wo;
  int y = (idx / Wo) % Ho;
  int c = (idx / (Wo * Ho)) % C;
  int b = idx / (Wo * Ho * C);
  const float* p = in + ((long)(b * C + c) * H + 2 * y) * W + 2 * x;
  out[idx] = fmaxf(fmaxf(p[0], p[1]), fmaxf(p[W], p[W + 1]));
}

__global__ void avgpool_kernel(const float* __restrict__ in, float* __restrict__ out)
{
  int idx = blockIdx.x * 256 + threadIdx.x;   // 64*256 = (b, c*4 + q)
  if (idx >= 64 * 256) return;
  int q = idx & 3, c = (idx >> 2) & 63, b = idx >> 8;
  int iy = q >> 1, ix = q & 1;
  const float* p = in + ((long)(b * 64 + c) * 16 + iy * 8) * 16 + ix * 8;
  float s = 0.f;
  for (int r = 0; r < 8; ++r)
    for (int col = 0; col < 8; ++col) s += p[r * 16 + col];
  out[idx] = s * (1.f / 64.f);
}

// ------------------------------ pointwise ----------------------------------
__global__ void glu_kernel(const float* __restrict__ in, float* __restrict__ out,
                           int Hh, int total)
{
  int idx = blockIdx.x * 256 + threadIdx.x;
  if (idx >= total) return;
  int h = idx % Hh, r = idx / Hh;
  float a = in[(long)r * 2 * Hh + h];
  float g = in[(long)r * 2 * Hh + Hh + h];
  out[idx] = a * (1.f / (1.f + expf(-g)));
}

__global__ void ln_kernel(const float* __restrict__ in, float* __restrict__ out, int L,
                          const float* __restrict__ g, const float* __restrict__ bta, int doRelu)
{
  int row = blockIdx.x;
  const float* r = in + (long)row * L;
  __shared__ float s1[256], s2[256];
  float sum = 0.f, ss = 0.f;
  for (int i = threadIdx.x; i < L; i += 256) { float v = r[i]; sum += v; ss += v * v; }
  s1[threadIdx.x] = sum; s2[threadIdx.x] = ss;
  __syncthreads();
  for (int w = 128; w > 0; w >>= 1) {
    if (threadIdx.x < w) { s1[threadIdx.x] += s1[threadIdx.x + w]; s2[threadIdx.x] += s2[threadIdx.x + w]; }
    __syncthreads();
  }
  float mean = s1[0] / L;
  float rstd = rsqrtf(s2[0] / L - mean * mean + 1e-5f);
  for (int i = threadIdx.x; i < L; i += 256) {
    float v = (r[i] - mean) * rstd * g[i] + bta[i];
    if (doRelu) v = fmaxf(v, 0.f);
    out[(long)row * L + i] = v;
  }
}

__global__ void init_act_trace_kernel(const float* __restrict__ sa, const float* __restrict__ st,
                                      float* __restrict__ act, float* __restrict__ trace)
{
  int idx = blockIdx.x * 256 + threadIdx.x;
  if (idx < 64 * 1024)  act[idx]   = sa[idx & 1023];
  if (idx < 64 * 32768) trace[idx] = st[idx & 32767];
}

__global__ void init_sync_kernel(const float* __restrict__ sa, const int* __restrict__ oL,
                                 const int* __restrict__ oR, float* __restrict__ ao,
                                 float* __restrict__ bo)
{
  int idx = blockIdx.x * 256 + threadIdx.x;
  if (idx >= 64 * 256) return;
  int j = idx & 255;
  ao[idx] = sa[oL[j]] * sa[oR[j]];
  bo[idx] = 1.f;
}

__global__ void concat_pre_kernel(const float* __restrict__ attn, const float* __restrict__ act,
                                  float* __restrict__ pre)
{
  int idx = blockIdx.x * 256 + threadIdx.x;
  if (idx >= 64 * 1536) return;
  int i = idx % 1536, b = idx / 1536;
  pre[idx] = (i < 512) ? attn[b * 512 + i] : act[b * 1024 + (i - 512)];
}

__global__ void trace_shift_kernel(float* __restrict__ trace, const float* __restrict__ state)
{
  int idx = blockIdx.x * 256 + threadIdx.x;   // (b,n)
  if (idx >= 64 * 1024) return;
  float* row = trace + (long)idx * 32;
  for (int i = 0; i < 31; ++i) row[i] = row[i + 1];
  row[31] = state[idx];
}

// act[b,n] = sum_h GLU(raw[b,n,h], raw[b,n,h+128]) * w2[n,h] + b2[n]
__global__ void nlm_dot_kernel(const float* __restrict__ raw, const float* __restrict__ w2,
                               const float* __restrict__ b2, float* __restrict__ act)
{
  int idx = blockIdx.x * 256 + threadIdx.x;   // b*1024 + n
  if (idx >= 64 * 1024) return;
  int n = idx & 1023, b = idx >> 10;
  const float* r = raw + (long)b * 262144 + (long)n * 256;
  const float* w = w2 + (long)n * 128;
  float s = b2[n];
  for (int h = 0; h < 128; ++h) {
    float a = r[h], g = r[h + 128];
    s += a * (1.f / (1.f + expf(-g))) * w[h];
  }
  act[idx] = s;
}

__global__ void sync_o_kernel(const float* __restrict__ act, const int* __restrict__ oL,
                              const int* __restrict__ oR, const float* __restrict__ dec_o,
                              float* __restrict__ ao, float* __restrict__ bo,
                              float* __restrict__ so)
{
  int idx = blockIdx.x * 256 + threadIdx.x;   // b*256 + j
  if (idx >= 64 * 256) return;
  int j = idx & 255, b = idx >> 8;
  float r = expf(-fminf(fmaxf(dec_o[j], 0.f), 15.f));
  float po = act[b * 1024 + oL[j]] * act[b * 1024 + oR[j]];
  float a  = r * ao[idx] + po;
  float bb = r * bo[idx] + 1.f;
  ao[idx] = a; bo[idx] = bb;
  so[idx] = a * rsqrtf(bb + 1e-8f);
}

// preds (B,OUT,T) + certs (B,2,T); one block per batch row.
__global__ void finalize_kernel(const float* __restrict__ pred, float* __restrict__ out, int t)
{
  int b = blockIdx.x;
  const float* r = pred + (long)b * 1000;
  __shared__ float sh[256];
  float mx = -3.4e38f;
  for (int i = threadIdx.x; i < 1000; i += 256) {
    float v = r[i];
    out[(long)b * 8000 + i * 8 + t] = v;
    mx = fmaxf(mx, v);
  }
  sh[threadIdx.x] = mx; __syncthreads();
  for (int w = 128; w > 0; w >>= 1) { if (threadIdx.x < w) sh[threadIdx.x] = fmaxf(sh[threadIdx.x], sh[threadIdx.x + w]); __syncthreads(); }
  float M = sh[0]; __syncthreads();
  float s = 0.f;
  for (int i = threadIdx.x; i < 1000; i += 256) s += expf(r[i] - M);
  sh[threadIdx.x] = s; __syncthreads();
  for (int w = 128; w > 0; w >>= 1) { if (threadIdx.x < w) sh[threadIdx.x] += sh[threadIdx.x + w]; __syncthreads(); }
  float Z = sh[0]; __syncthreads();
  float e = 0.f;
  for (int i = threadIdx.x; i < 1000; i += 256) {
    float p = expf(r[i] - M) / Z;
    e -= p * logf(p + 1e-10f);
  }
  sh[threadIdx.x] = e; __syncthreads();
  for (int w = 128; w > 0; w >>= 1) { if (threadIdx.x < w) sh[threadIdx.x] += sh[threadIdx.x + w]; __syncthreads(); }
  if (threadIdx.x == 0) {
    float ne = sh[0] * 0.14476482730108395f;     // 1/ln(1000)
    out[512000 + b * 16 + t]     = ne;
    out[512000 + b * 16 + 8 + t] = 1.f - ne;
  }
}

// ---------------------------------------------------------------------------
extern "C" void kernel_launch(void* const* d_in, const int* in_sizes, int n_in,
                              void* d_out, int out_size, void* d_ws, size_t ws_size,
                              hipStream_t stream)
{
  (void)in_sizes; (void)n_in; (void)out_size; (void)ws_size;
  const float* x      = (const float*)d_in[0];
  const float* c1w    = (const float*)d_in[1];  const float* c1b  = (const float*)d_in[2];
  const float* bn1g   = (const float*)d_in[3];  const float* bn1b = (const float*)d_in[4];
  const float* c2w    = (const float*)d_in[5];  const float* c2b  = (const float*)d_in[6];
  const float* bn2g   = (const float*)d_in[7];  const float* bn2b = (const float*)d_in[8];
  const float* c3w    = (const float*)d_in[9];  const float* c3b  = (const float*)d_in[10];
  const float* bn3g   = (const float*)d_in[11]; const float* bn3b = (const float*)d_in[12];
  const float* c4w    = (const float*)d_in[13]; const float* c4b  = (const float*)d_in[14];
  const float* bn4g   = (const float*)d_in[15]; const float* bn4b = (const float*)d_in[16];
  const float* fc1w   = (const float*)d_in[17]; const float* fc1b = (const float*)d_in[18];
  const float* fc2w   = (const float*)d_in[19]; const float* fc2b = (const float*)d_in[20];
  const float* kvw    = (const float*)d_in[21]; const float* kvb  = (const float*)d_in[22];
  const float* kvlg   = (const float*)d_in[23]; const float* kvlb = (const float*)d_in[24];
  const float* synw   = (const float*)d_in[25]; const float* synb = (const float*)d_in[26];
  const float* synlg  = (const float*)d_in[27]; const float* synlb= (const float*)d_in[28];
  const float* nlm_w1 = (const float*)d_in[29]; const float* nlm_b1 = (const float*)d_in[30];
  const float* nlm_w2 = (const float*)d_in[31]; const float* nlm_b2 = (const float*)d_in[32];
  const float* start_act   = (const float*)d_in[33];
  const float* start_trace = (const float*)d_in[34];
  const float* dec_o  = (const float*)d_in[36];
  // d_in[35]=dec_a, d_in[37]=qw, d_in[38]=qb, d_in[53]=aL, d_in[54]=aR are dead:
  // softmax over a single key is exactly 1, so the query path has no effect.
  const float* attn_inw = (const float*)d_in[39];
  const float* attn_inb = (const float*)d_in[40];
  const float* attn_ow  = (const float*)d_in[41];
  const float* attn_ob  = (const float*)d_in[42];
  const float* op_w1  = (const float*)d_in[43]; const float* op_b1 = (const float*)d_in[44];
  const float* op_l1g = (const float*)d_in[45]; const float* op_l1b= (const float*)d_in[46];
  const float* op_w2  = (const float*)d_in[47]; const float* op_b2 = (const float*)d_in[48];
  const float* op_l2g = (const float*)d_in[49]; const float* op_l2b= (const float*)d_in[50];
  const float* op_w3  = (const float*)d_in[51]; const float* op_b3 = (const float*)d_in[52];
  const int* oLp = (const int*)d_in[55];
  const int* oRp = (const int*)d_in[56];
  float* out = (float*)d_out;

  // --- workspace bump allocator -------------------------------------------
  char* wsp = (char*)d_ws;
  size_t off = 0;
  auto take = [&](size_t bytes) -> char* {
    char* p = wsp + off;
    off += (bytes + 255) & ~(size_t)255;
    return p;
  };
  float* bufA   = (float*)take(64UL * 32 * 64 * 64 * 4);   // 32 MB conv scratch
  float* bufB   = (float*)take(64UL * 32 * 64 * 64 * 4);   // 32 MB conv scratch
  float* nlmraw = bufA;                                     // 64 MB, reused after encoder
  float* stats  = (float*)take(2UL * 64 * 4);
  float* pooled = (float*)take(64UL * 256 * 4);
  float* fc1raw = (float*)take(64UL * 256 * 4);
  float* glu1   = (float*)take(64UL * 128 * 4);
  float* enc    = (float*)take(64UL * 512 * 4);
  float* kvraw  = (float*)take(64UL * 512 * 4);
  float* kvbuf  = (float*)take(64UL * 512 * 4);
  float* vh     = (float*)take(64UL * 512 * 4);
  float* attn   = (float*)take(64UL * 512 * 4);
  float* act    = (float*)take(64UL * 1024 * 4);
  float* trace  = (float*)take(64UL * 1024 * 32 * 4);
  float* ao     = (float*)take(64UL * 256 * 4);
  float* bo     = (float*)take(64UL * 256 * 4);
  float* so     = (float*)take(64UL * 256 * 4);
  float* pre    = (float*)take(64UL * 1536 * 4);
  float* synraw = (float*)take(64UL * 2048 * 4);
  float* state  = (float*)take(64UL * 1024 * 4);
  float* o1     = (float*)take(64UL * 256 * 4);
  float* h1     = (float*)take(64UL * 256 * 4);
  float* o2     = (float*)take(64UL * 64 * 4);
  float* h2     = (float*)take(64UL * 64 * 4);
  float* pred   = (float*)take(64UL * 1000 * 4);
  unsigned short* fc1P = (unsigned short*)take(8UL * 16 * 512 * 2);
  unsigned short* fc2P = (unsigned short*)take(4UL * 32 * 512 * 2);
  unsigned short* kvP  = (unsigned short*)take(16UL * 32 * 512 * 2);
  unsigned short* wvP  = (unsigned short*)take(16UL * 32 * 512 * 2);
  unsigned short* aowP = (unsigned short*)take(16UL * 32 * 512 * 2);
  unsigned short* synP = (unsigned short*)take(48UL * 128 * 512 * 2);
  unsigned short* nlmP = (unsigned short*)take(1024UL * 16 * 512 * 2);
  unsigned short* op1P = (unsigned short*)take(8UL * 16 * 512 * 2);
  unsigned short* op2P = (unsigned short*)take(8UL * 4 * 512 * 2);
  unsigned short* op3P = (unsigned short*)take(2UL * 64 * 512 * 2);  // NT padded to 64

  auto G1 = [](long n) { return dim3((unsigned)((n + 255) / 256)); };

  // --- one-time weight repack to bf16 WMMA fragments ----------------------
  pack_b_kernel<<<G1(8 * 16 * 32), 256, 0, stream>>>(fc1w, 0, 256, 256, 256, 16, fc1P, 0);
  pack_b_kernel<<<G1(4 * 32 * 32), 256, 0, stream>>>(fc2w, 0, 512, 128, 512, 32, fc2P, 0);
  pack_b_kernel<<<G1(16 * 32 * 32), 256, 0, stream>>>(kvw, 0, 512, 512, 512, 32, kvP, 0);
  pack_b_kernel<<<G1(16 * 32 * 32), 256, 0, stream>>>(attn_inw + 1024, 0, 1536, 512, 512, 32, wvP, 0);
  pack_b_kernel<<<G1(16 * 32 * 32), 256, 0, stream>>>(attn_ow, 0, 512, 512, 512, 32, aowP, 0);
  pack_b_kernel<<<G1(48 * 128 * 32), 256, 0, stream>>>(synw, 0, 2048, 1536, 2048, 128, synP, 0);
  pack_b_kernel<<<dim3((unsigned)((1 * 16 * 32 + 255) / 256), 1024), 256, 0, stream>>>(
      nlm_w1, 8192, 256, 32, 256, 16, nlmP, 8192);
  pack_b_kernel<<<G1(8 * 16 * 32), 256, 0, stream>>>(op_w1, 0, 256, 256, 256, 16, op1P, 0);
  pack_b_kernel<<<G1(8 * 4 * 32), 256, 0, stream>>>(op_w2, 0, 64, 256, 64, 4, op2P, 0);
  pack_b_kernel<<<G1(2 * 64 * 32), 256, 0, stream>>>(op_w3, 0, 1000, 64, 1000, 64, op3P, 0);

  // --- CNN encoder --------------------------------------------------------
  conv3x3_kernel<<<G1(4194304), 256, 0, stream>>>(x, c1w, c1b, bufA, 3, 16, 64, 64, 4194304);
  bn_stats_kernel<<<16, 256, 0, stream>>>(bufA, 16, 4096, bn1g, bn1b, stats);
  bn_apply_relu_kernel<<<G1(4194304), 256, 0, stream>>>(bufA, stats, 16, 4096, 4194304);
  conv3x3_kernel<<<G1(8388608), 256, 0, stream>>>(bufA, c2w, c2b, bufB, 16, 32, 64, 64, 8388608);
  bn_stats_kernel<<<32, 256, 0, stream>>>(bufB, 32, 4096, bn2g, bn2b, stats);
  bn_apply_relu_kernel<<<G1(8388608), 256, 0, stream>>>(bufB, stats, 32, 4096, 8388608);
  maxpool2_kernel<<<G1(2097152), 256, 0, stream>>>(bufB, bufA, 32, 64, 64, 2097152);
  conv3x3_kernel<<<G1(4194304), 256, 0, stream>>>(bufA, c3w, c3b, bufB, 32, 64, 32, 32, 4194304);
  bn_stats_kernel<<<64, 256, 0, stream>>>(bufB, 64, 1024, bn3g, bn3b, stats);
  bn_apply_relu_kernel<<<G1(4194304), 256, 0, stream>>>(bufB, stats, 64, 1024, 4194304);
  maxpool2_kernel<<<G1(1048576), 256, 0, stream>>>(bufB, bufA, 64, 32, 32, 1048576);
  conv3x3_kernel<<<G1(1048576), 256, 0, stream>>>(bufA, c4w, c4b, bufB, 64, 64, 16, 16, 1048576);
  bn_stats_kernel<<<64, 256, 0, stream>>>(bufB, 64, 256, bn4g, bn4b, stats);
  bn_apply_relu_kernel<<<G1(1048576), 256, 0, stream>>>(bufB, stats, 64, 256, 1048576);
  avgpool_kernel<<<G1(16384), 256, 0, stream>>>(bufB, pooled);

  gemm_bf16_wmma<<<dim3(4, 4, 1), 32, 0, stream>>>(pooled, 256, 0, fc1P, 0, 8, 16,
                                                   fc1raw, 256, 0, fc1b, 0, 256);
  glu_kernel<<<G1(8192), 256, 0, stream>>>(fc1raw, glu1, 128, 8192);
  gemm_bf16_wmma<<<dim3(8, 4, 1), 32, 0, stream>>>(glu1, 128, 0, fc2P, 0, 4, 32,
                                                   enc, 512, 0, fc2b, 0, 512);
  gemm_bf16_wmma<<<dim3(8, 4, 1), 32, 0, stream>>>(enc, 512, 0, kvP, 0, 16, 32,
                                                   kvraw, 512, 0, kvb, 0, 512);
  ln_kernel<<<64, 256, 0, stream>>>(kvraw, kvbuf, 512, kvlg, kvlb, 0);
  // single-KV attention collapses exactly (softmax over one key == 1):
  gemm_bf16_wmma<<<dim3(8, 4, 1), 32, 0, stream>>>(kvbuf, 512, 0, wvP, 0, 16, 32,
                                                   vh, 512, 0, attn_inb + 1024, 0, 512);
  gemm_bf16_wmma<<<dim3(8, 4, 1), 32, 0, stream>>>(vh, 512, 0, aowP, 0, 16, 32,
                                                   attn, 512, 0, attn_ob, 0, 512);

  // --- recurrent state init ----------------------------------------------
  init_act_trace_kernel<<<G1(2097152), 256, 0, stream>>>(start_act, start_trace, act, trace);
  init_sync_kernel<<<G1(16384), 256, 0, stream>>>(start_act, oLp, oRp, ao, bo);

  // --- T = 8 ticks --------------------------------------------------------
  for (int t = 0; t < 8; ++t) {
    concat_pre_kernel<<<G1(64 * 1536), 256, 0, stream>>>(attn, act, pre);
    gemm_bf16_wmma<<<dim3(32, 4, 1), 32, 0, stream>>>(pre, 1536, 0, synP, 0, 48, 128,
                                                      synraw, 2048, 0, synb, 0, 2048);
    glu_kernel<<<G1(65536), 256, 0, stream>>>(synraw, state, 1024, 65536);
    ln_kernel<<<64, 256, 0, stream>>>(state, state, 1024, synlg, synlb, 0);
    trace_shift_kernel<<<G1(65536), 256, 0, stream>>>(trace, state);
    // per-neuron MLPs: 1024 batched (64x32)@(32x256) GEMMs, 4 WMMAs per wave
    gemm_bf16_wmma<<<dim3(4, 4, 1024), 32, 0, stream>>>(trace, 32768, 32, nlmP, 8192, 1, 16,
                                                        nlmraw, 262144, 256, nlm_b1, 256, 256);
    nlm_dot_kernel<<<G1(65536), 256, 0, stream>>>(nlmraw, nlm_w2, nlm_b2, act);
    sync_o_kernel<<<G1(16384), 256, 0, stream>>>(act, oLp, oRp, dec_o, ao, bo, so);
    gemm_bf16_wmma<<<dim3(4, 4, 1), 32, 0, stream>>>(so, 256, 0, op1P, 0, 8, 16,
                                                     o1, 256, 0, op_b1, 0, 256);
    ln_kernel<<<64, 256, 0, stream>>>(o1, h1, 256, op_l1g, op_l1b, 1);
    gemm_bf16_wmma<<<dim3(1, 4, 1), 32, 0, stream>>>(h1, 256, 0, op2P, 0, 8, 4,
                                                     o2, 64, 0, op_b2, 0, 64);
    ln_kernel<<<64, 256, 0, stream>>>(o2, h2, 64, op_l2g, op_l2b, 1);
    gemm_bf16_wmma<<<dim3(16, 4, 1), 32, 0, stream>>>(h2, 64, 0, op3P, 0, 2, 64,
                                                      pred, 1000, 0, op_b3, 0, 1000);
    finalize_kernel<<<64, 256, 0, stream>>>(pred, out, t);
  }
}
